// SpatialBlock_41231686041881
// MI455X (gfx1250) — compile-verified, matrix operand order checked by reading, not run on hardware
//
#include <hip/hip_runtime.h>
#include <hip/hip_bf16.h>

typedef __attribute__((ext_vector_type(16))) _Float16 v16h;
typedef __attribute__((ext_vector_type(8)))  float    v8f;
typedef __attribute__((ext_vector_type(4)))  int      v4i;

#define B_    16
#define Hh    56
#define Ww    56
#define Ls    3136          // Hh*Ww
#define Cc    384
#define Mrows 50176         // B_*Ls
#define NHh   12
#define HD    32
#define NTOK  49            // 7*7
#define QKVN  1152          // 3*Cc
#define HID   1536          // 4*Cc
#define LDA   40            // padded f16 row stride for 32-wide K tiles (80B, 16B-aligned)

union FragU { unsigned int u[8]; v16h v; };

// K index held by VGPR j for a lane in half (0: lanes 0-15, 1: lanes 16-31)
// per the CDNA5 16-bit A/B fragment layout (16x32 / 32x16, K pairs per VGPR).
__device__ __forceinline__ int kidx_of(int j, int half) {
  return (j < 4) ? (half * 8 + 2 * j) : (16 + half * 8 + 2 * (j - 4));
}

// ---- 16-byte global->LDS copy: async DMA path (ASYNCcnt) if available -------
#if defined(__has_builtin)
#if __has_builtin(__builtin_amdgcn_global_load_async_to_lds_b128)
#define HAS_ASYNC_CP 1
#endif
#endif
#ifndef HAS_ASYNC_CP
#define HAS_ASYNC_CP 0
#endif

__device__ __forceinline__ void cp16B(_Float16* l, const _Float16* g) {
#if HAS_ASYNC_CP
  __builtin_amdgcn_global_load_async_to_lds_b128(
      (__attribute__((address_space(1))) v4i*)g,
      (__attribute__((address_space(3))) v4i*)l, 0, 0);
#else
  *(float4*)l = *(const float4*)g;
#endif
}
__device__ __forceinline__ void cp_wait_all() {
#if HAS_ASYNC_CP
  asm volatile("s_wait_asynccnt 0" ::: "memory");
#endif
}

// ---------------------------------------------------------------- cvt f32->f16
__global__ void cvt_f32_to_f16(const float* __restrict__ in,
                               _Float16* __restrict__ out, int n) {
  int i = blockIdx.x * 256 + threadIdx.x;
  if (i < n) out[i] = (_Float16)in[i];
}
// transpose-convert: in[K][N] (f32) -> out[N][K] (f16); writes coalesced along K
__global__ void cvt_f32_to_f16_T(const float* __restrict__ in,
                                 _Float16* __restrict__ out, int K, int N) {
  int idx = blockIdx.x * 256 + threadIdx.x;
  if (idx < K * N) {
    int n = idx / K, k = idx % K;
    out[idx] = (_Float16)in[(size_t)k * N + n];
  }
}

// ------------------------------------------------- depthwise 3x3 CPE + residual
__global__ void dwconv_kernel(const float* __restrict__ x,
                              const float* __restrict__ w9,
                              const float* __restrict__ bias,
                              float* __restrict__ out) {
  size_t idx = (size_t)blockIdx.x * 256 + threadIdx.x;
  if (idx >= (size_t)Mrows * Cc) return;
  int c = (int)(idx % Cc);
  size_t rl = idx / Cc;
  int l = (int)(rl % Ls);
  int b = (int)(rl / Ls);
  int h = l / Ww, w0 = l % Ww;
  float acc = bias[c];
#pragma unroll
  for (int dh = -1; dh <= 1; ++dh)
#pragma unroll
    for (int dw = -1; dw <= 1; ++dw) {
      int hh = h + dh, ww = w0 + dw;
      if (hh >= 0 && hh < Hh && ww >= 0 && ww < Ww)
        acc += x[((size_t)b * Ls + hh * Ww + ww) * Cc + c] *
               w9[c * 9 + (dh + 1) * 3 + (dw + 1)];
    }
  out[idx] = x[idx] + acc;
}

// ------------------------------------------------------------------ layer norm
__global__ __launch_bounds__(128) void ln_kernel(const float* __restrict__ x,
                                                 const float* __restrict__ g,
                                                 const float* __restrict__ bb,
                                                 _Float16* __restrict__ out) {
  __shared__ float red[128];
  const int row = blockIdx.x, tid = threadIdx.x;
  const float* xr = x + (size_t)row * Cc;
  float s = 0.f;
  for (int i = tid; i < Cc; i += 128) s += xr[i];
  red[tid] = s; __syncthreads();
  for (int off = 64; off > 0; off >>= 1) {
    if (tid < off) red[tid] += red[tid + off];
    __syncthreads();
  }
  float mean = red[0] * (1.0f / Cc);
  __syncthreads();
  float v = 0.f;
  for (int i = tid; i < Cc; i += 128) { float d = xr[i] - mean; v += d * d; }
  red[tid] = v; __syncthreads();
  for (int off = 64; off > 0; off >>= 1) {
    if (tid < off) red[tid] += red[tid + off];
    __syncthreads();
  }
  float rstd = rsqrtf(red[0] * (1.0f / Cc) + 1e-5f);
  _Float16* orow = out + (size_t)row * Cc;
  for (int i = tid; i < Cc; i += 128)
    orow[i] = (_Float16)((xr[i] - mean) * rstd * g[i] + bb[i]);
}

// --------------------------------------------------------- WMMA tiled GEMM
// C[M,N] = A[M,K](f16) @ Bt[N,K](f16, pre-transposed) + bias ; modes:
//  0: store f16   1: GELU + store f16   2: +resid(f32) store f32
// 128 thr (4 waves), tile 64x64, K step 32, double-buffered async LDS staging.
__global__ __launch_bounds__(128) void gemm_wmma_kernel(
    const _Float16* __restrict__ A, const _Float16* __restrict__ Bt,
    const float* __restrict__ bias, const float* __restrict__ resid,
    void* __restrict__ outp, int Ndim, int Kdim, int mode) {
  __shared__ __align__(16) _Float16 lds[4 * 64 * LDA];  // [A0][A1][B0][B1]
  const int tid = threadIdx.x;
  const int wave = tid >> 5, lane = tid & 31, half = lane >> 4, l16 = lane & 15;
  const int m0 = blockIdx.y * 64, n0 = blockIdx.x * 64;
  const int TS = 64 * LDA;   // one tile buffer, in f16 elements

  // staging: 256 16B segments per matrix; thread owns (tid) and (tid+128)
  const int sr = tid >> 2, sg = tid & 3;     // row 0..31 , segment 0..3
  const _Float16* gA = A  + (size_t)(m0 + sr) * Kdim + sg * 8;
  const _Float16* gB = Bt + (size_t)(n0 + sr) * Kdim + sg * 8;
  const size_t rowAdv = (size_t)32 * Kdim;   // rows 32..63
  const int lo = sr * LDA + sg * 8;

  v8f acc[4];
  { v8f z = {}; acc[0] = z; acc[1] = z; acc[2] = z; acc[3] = z; }

  const int KT = Kdim >> 5;
#define STAGE(bufi, kt)                                                \
  do {                                                                 \
    const size_t ko = (size_t)(kt) * 32;                               \
    cp16B(lds + (bufi)*TS + lo,                gA + ko);               \
    cp16B(lds + (bufi)*TS + lo + 32*LDA,       gA + ko + rowAdv);      \
    cp16B(lds + (2+(bufi))*TS + lo,            gB + ko);               \
    cp16B(lds + (2+(bufi))*TS + lo + 32*LDA,   gB + ko + rowAdv);      \
  } while (0)

  STAGE(0, 0);
  for (int kt = 0; kt < KT; ++kt) {
    const int buf = kt & 1;
    cp_wait_all();
    __syncthreads();                       // tile kt visible to all waves
    if (kt + 1 < KT) STAGE(buf ^ 1, kt + 1);  // overlap next DMA with compute

    const _Float16* tA = lds + buf * TS;
    const _Float16* tB = lds + (2 + buf) * TS;
    FragU a;
#pragma unroll
    for (int j = 0; j < 8; ++j)
      a.u[j] = *(const unsigned int*)&tA[(wave * 16 + l16) * LDA + kidx_of(j, half)];
    FragU bf[4];
#pragma unroll
    for (int nt = 0; nt < 4; ++nt)
#pragma unroll
      for (int j = 0; j < 8; ++j)
        bf[nt].u[j] = *(const unsigned int*)&tB[(nt * 16 + l16) * LDA + kidx_of(j, half)];
#pragma unroll
    for (int nt = 0; nt < 4; ++nt)
      acc[nt] = __builtin_amdgcn_wmma_f32_16x16x32_f16(
          false, a.v, false, bf[nt].v, (short)0, acc[nt], false, false);
    __syncthreads();                       // done reading buf before its reuse
  }
#undef STAGE

#pragma unroll
  for (int nt = 0; nt < 4; ++nt) {
    int gn = n0 + nt * 16 + l16;
    float bv = bias ? bias[gn] : 0.0f;
#pragma unroll
    for (int r = 0; r < 8; ++r) {
      int gm = m0 + wave * 16 + half * 8 + r;
      float v = acc[nt][r] + bv;
      size_t o = (size_t)gm * Ndim + gn;
      if (mode == 2) {
        ((float*)outp)[o] = v + resid[o];
      } else {
        if (mode == 1) v = 0.5f * v * (1.0f + erff(v * 0.70710678118f));
        ((_Float16*)outp)[o] = (_Float16)v;
      }
    }
  }
}

// ---------------------------------------------- windowed attention (one wh/blk)
__global__ __launch_bounds__(128) void attn_wmma_kernel(
    const _Float16* __restrict__ qkv, _Float16* __restrict__ o16) {
  __shared__ __align__(16) _Float16 q16[64 * LDA];
  __shared__ __align__(16) _Float16 k16[64 * LDA];
  __shared__ _Float16 vv16[32 * 66];   // [ch][tok]
  __shared__ float    sbuf[64 * 66];
  __shared__ _Float16 p16[64 * 66];

  const int tid = threadIdx.x;
  const int wave = tid >> 5, lane = tid & 31, half = lane >> 4, l16 = lane & 15;
  const int wh = blockIdx.x;
  const int win = wh / NHh, head = wh % NHh;
  const int b = win >> 6, widx = win & 63, wy = widx >> 3, wx = widx & 7;

  auto rowOf = [&](int t) -> size_t {
    int ty = t / 7, tx = t % 7;
    return (size_t)b * Ls + (size_t)(wy * 7 + ty) * Ww + (wx * 7 + tx);
  };
  const int qoff = head * HD, koff = Cc + head * HD, voff = 2 * Cc + head * HD;

  // q,k rows: 16B segments (4 per token), pad tokens zero-filled
#pragma unroll
  for (int part = 0; part < 2; ++part) {
    int idx = tid + part * 128;              // 0..255
    int t = idx >> 2, sg = idx & 3;
    _Float16* lq = q16 + t * LDA + sg * 8;
    _Float16* lk = k16 + t * LDA + sg * 8;
    if (t < NTOK) {
      size_t rb = rowOf(t) * QKVN;
      cp16B(lq, qkv + rb + qoff + sg * 8);
      cp16B(lk, qkv + rb + koff + sg * 8);
    } else {
      float4 z = {};
      *(float4*)lq = z;
      *(float4*)lk = z;
    }
  }
  // v transposed [ch][tok] (scalar scatter)
#pragma unroll
  for (int i = 0; i < 16; ++i) {
    int e = tid + i * 128;                   // 0..2047
    int t = e >> 5, ch = e & 31;
    _Float16 val = (_Float16)0.0f;
    if (t < NTOK) val = qkv[rowOf(t) * QKVN + voff + ch];
    vv16[ch * 66 + t] = val;
  }
  cp_wait_all();
  __syncthreads();

  // S = Q @ K^T  (M=64, N=64, K=32)
  {
    FragU a;
#pragma unroll
    for (int j = 0; j < 8; ++j)
      a.u[j] = *(const unsigned int*)&q16[(wave * 16 + l16) * LDA + kidx_of(j, half)];
    FragU bf[4];
#pragma unroll
    for (int nt = 0; nt < 4; ++nt)
#pragma unroll
      for (int j = 0; j < 8; ++j)
        bf[nt].u[j] = *(const unsigned int*)&k16[(nt * 16 + l16) * LDA + kidx_of(j, half)];
#pragma unroll
    for (int nt = 0; nt < 4; ++nt) {
      v8f z = {};
      v8f s = __builtin_amdgcn_wmma_f32_16x16x32_f16(
          false, a.v, false, bf[nt].v, (short)0, z, false, false);
#pragma unroll
      for (int r = 0; r < 8; ++r)
        sbuf[(wave * 16 + half * 8 + r) * 66 + nt * 16 + l16] = s[r];
    }
  }
  __syncthreads();

  // softmax per valid row (scale applied here), pad -> 0
  if (tid < 64) {
    int t = tid;
    if (t < NTOK) {
      const float scale = 0.1767766952966369f;   // 1/sqrt(32)
      float mx = -1e30f;
      for (int n = 0; n < NTOK; ++n) mx = fmaxf(mx, sbuf[t * 66 + n] * scale);
      float sum = 0.f;
      for (int n = 0; n < NTOK; ++n) {
        float e = __expf(sbuf[t * 66 + n] * scale - mx);
        sbuf[t * 66 + n] = e;
        sum += e;
      }
      float inv = 1.0f / sum;
      for (int n = 0; n < 64; ++n)
        p16[t * 66 + n] = (n < NTOK) ? (_Float16)(sbuf[t * 66 + n] * inv)
                                     : (_Float16)0.0f;
    } else {
      for (int n = 0; n < 64; ++n) p16[t * 66 + n] = (_Float16)0.0f;
    }
  }
  __syncthreads();

  // O = P @ V  (M=64, N=32, K=64 in two 32-steps)
  v8f accO[2];
  { v8f z = {}; accO[0] = z; accO[1] = z; }
#pragma unroll
  for (int ks = 0; ks < 2; ++ks) {
    FragU a;
#pragma unroll
    for (int j = 0; j < 8; ++j)
      a.u[j] = *(const unsigned int*)&p16[(wave * 16 + l16) * 66 + ks * 32 + kidx_of(j, half)];
    FragU bf[2];
#pragma unroll
    for (int nt = 0; nt < 2; ++nt)
#pragma unroll
      for (int j = 0; j < 8; ++j)
        bf[nt].u[j] = *(const unsigned int*)&vv16[(nt * 16 + l16) * 66 + ks * 32 + kidx_of(j, half)];
#pragma unroll
    for (int nt = 0; nt < 2; ++nt)
      accO[nt] = __builtin_amdgcn_wmma_f32_16x16x32_f16(
          false, a.v, false, bf[nt].v, (short)0, accO[nt], false, false);
  }
  // scatter back to natural (b,l,c) layout
#pragma unroll
  for (int nt = 0; nt < 2; ++nt) {
#pragma unroll
    for (int r = 0; r < 8; ++r) {
      int m = wave * 16 + half * 8 + r;
      if (m < NTOK)
        o16[rowOf(m) * Cc + head * HD + nt * 16 + l16] = (_Float16)accO[nt][r];
    }
  }
}

// ----------------------------------------------------------------- launch
extern "C" void kernel_launch(void* const* d_in, const int* in_sizes, int n_in,
                              void* d_out, int out_size, void* d_ws, size_t ws_size,
                              hipStream_t stream) {
  (void)in_sizes; (void)n_in; (void)out_size; (void)ws_size;
  const float* x      = (const float*)d_in[0];
  const float* cpe0_w = (const float*)d_in[1];
  const float* cpe0_b = (const float*)d_in[2];
  const float* cpe1_w = (const float*)d_in[3];
  const float* cpe1_b = (const float*)d_in[4];
  const float* n1g    = (const float*)d_in[5];
  const float* n1b    = (const float*)d_in[6];
  const float* qkv_w  = (const float*)d_in[7];
  const float* qkv_b  = (const float*)d_in[8];
  const float* proj_w = (const float*)d_in[9];
  const float* proj_b = (const float*)d_in[10];
  const float* n2g    = (const float*)d_in[11];
  const float* n2b    = (const float*)d_in[12];
  const float* fc1_w  = (const float*)d_in[13];
  const float* fc1_b  = (const float*)d_in[14];
  const float* fc2_w  = (const float*)d_in[15];
  const float* fc2_b  = (const float*)d_in[16];

  char* ws = (char*)d_ws;
  size_t off = 0;
  _Float16* qkvw16  = (_Float16*)(ws + off); off += (size_t)Cc * QKVN * 2;  // [N][K]
  _Float16* projw16 = (_Float16*)(ws + off); off += (size_t)Cc * Cc * 2;
  _Float16* fc1w16  = (_Float16*)(ws + off); off += (size_t)Cc * HID * 2;
  _Float16* fc2w16  = (_Float16*)(ws + off); off += (size_t)HID * Cc * 2;
  off = (off + 255) & ~(size_t)255;
  float*    bufA  = (float*)(ws + off);    off += (size_t)Mrows * Cc * 4;  // shortcut / x3
  float*    x2    = (float*)(ws + off);    off += (size_t)Mrows * Cc * 4;
  _Float16* buf16 = (_Float16*)(ws + off); off += (size_t)Mrows * Cc * 2;  // LN out
  _Float16* qkv16 = (_Float16*)(ws + off);                                  // [qkv|o] region
  _Float16* hbuf16 = qkv16;   // fc1 output reuses qkv16+o16 (exactly Mrows*HID f16)
  off += (size_t)Mrows * QKVN * 2;
  _Float16* o16 = (_Float16*)(ws + off);   off += (size_t)Mrows * Cc * 2;

  // 1. weight conversion f32 -> f16, transposed to [N][K]
  cvt_f32_to_f16_T<<<(Cc * QKVN + 255) / 256, 256, 0, stream>>>(qkv_w, qkvw16, Cc, QKVN);
  cvt_f32_to_f16_T<<<(Cc * Cc + 255) / 256, 256, 0, stream>>>(proj_w, projw16, Cc, Cc);
  cvt_f32_to_f16_T<<<(Cc * HID + 255) / 256, 256, 0, stream>>>(fc1_w, fc1w16, Cc, HID);
  cvt_f32_to_f16_T<<<(HID * Cc + 255) / 256, 256, 0, stream>>>(fc2_w, fc2w16, HID, Cc);

  const size_t tot = (size_t)Mrows * Cc;
  // 2. CPE0 -> shortcut
  dwconv_kernel<<<(unsigned)((tot + 255) / 256), 256, 0, stream>>>(x, cpe0_w, cpe0_b, bufA);
  // 3. LN1 -> f16
  ln_kernel<<<Mrows, 128, 0, stream>>>(bufA, n1g, n1b, buf16);
  // 4. QKV GEMM (50176 x 1152 x 384)
  gemm_wmma_kernel<<<dim3(QKVN / 64, Mrows / 64), 128, 0, stream>>>(
      buf16, qkvw16, qkv_b, nullptr, qkv16, QKVN, Cc, 0);
  // 5. windowed attention (1024 windows x 12 heads)
  attn_wmma_kernel<<<1024 * NHh, 128, 0, stream>>>(qkv16, o16);
  // 6. proj GEMM + shortcut residual -> x2 (f32)
  gemm_wmma_kernel<<<dim3(Cc / 64, Mrows / 64), 128, 0, stream>>>(
      o16, projw16, proj_b, bufA, x2, Cc, Cc, 2);
  // 7. CPE1 -> bufA (x3)
  dwconv_kernel<<<(unsigned)((tot + 255) / 256), 256, 0, stream>>>(x2, cpe1_w, cpe1_b, bufA);
  // 8. LN2 -> f16
  ln_kernel<<<Mrows, 128, 0, stream>>>(bufA, n2g, n2b, buf16);
  // 9. FC1 GEMM + GELU (50176 x 1536 x 384) -> hidden f16
  gemm_wmma_kernel<<<dim3(HID / 64, Mrows / 64), 128, 0, stream>>>(
      buf16, fc1w16, fc1_b, nullptr, hbuf16, HID, Cc, 1);
  // 10. FC2 GEMM + x3 residual (50176 x 384 x 1536) -> d_out f32
  gemm_wmma_kernel<<<dim3(Cc / 64, Mrows / 64), 128, 0, stream>>>(
      hbuf16, fc2w16, fc2_b, bufA, (float*)d_out, Cc, HID, 2);
}
